// _gcn_43765716747105
// MI455X (gfx1250) — matrix operand order repaired and checked by reading, not compiled, not run on hardware
//
#include <hip/hip_runtime.h>
#include <hip/hip_bf16.h>
#include <cstdint>

typedef __attribute__((ext_vector_type(16))) __bf16 v16bf;
typedef __attribute__((ext_vector_type(8)))  float  v8f;

union FragB16 { v16bf v; uint4 q[2]; };

// ---------------------------------------------------------------------------
// CDNA5 async global->LDS copy (16B per lane), tracked by ASYNCcnt.
// LDS offset = low 32 bits of flat pointer (ISA: LDS aperture uses addr[31:0]).
// ---------------------------------------------------------------------------
__device__ __forceinline__ void async_copy16(const __bf16* g, const __bf16* l) {
    unsigned      lds  = (unsigned)(uintptr_t)l;
    unsigned long long ga = (unsigned long long)(uintptr_t)g;
    asm volatile("global_load_async_to_lds_b128 %0, %1, off"
                 :: "v"(lds), "v"(ga) : "memory");
}

__device__ __forceinline__ void wait_async0() {
#if defined(__has_builtin) && __has_builtin(__builtin_amdgcn_s_wait_asynccnt)
    __builtin_amdgcn_s_wait_asynccnt(0);
#else
    asm volatile("s_wait_asynccnt 0x0" ::: "memory");
#endif
}

// ---------------------------------------------------------------------------
// f32 -> bf16 convert (element counts are multiples of 4)
// ---------------------------------------------------------------------------
__global__ void k_cvt_bf16(const float* __restrict__ in, __bf16* __restrict__ out, int n) {
    int i = (blockIdx.x * blockDim.x + threadIdx.x) * 4;
    if (i >= n) return;
    float4 f = *(const float4*)(in + i);
    union { __bf16 h[4]; uint2 u; } r;
    r.h[0] = (__bf16)f.x; r.h[1] = (__bf16)f.y;
    r.h[2] = (__bf16)f.z; r.h[3] = (__bf16)f.w;
    *(uint2*)(out + i) = r.u;
}

// ---------------------------------------------------------------------------
// WMMA bf16 GEMM:  C[M,N] = A[M,K] * B[K,N]
//   A row-major (lda=K), B column-major (ldb=K), C column-major (ldc=M).
//   K % 64 == 0, M % BM == 0, N % BN == 0.
// Wave tile = 32 x (16*TN).  Double-buffered LDS K-tiles of 64, filled with
// global_load_async_to_lds_b128 and synchronized via s_wait_asynccnt.
// ---------------------------------------------------------------------------
template <int BM, int BN, int TN>
__global__ __launch_bounds__((BM / 32) * (BN / (16 * TN)) * 32)
void k_gemm_bf16(const __bf16* __restrict__ A, int lda,
                 const __bf16* __restrict__ B, int ldb,
                 float* __restrict__ C, int ldc, int K) {
    constexpr int WM   = BM / 32;
    constexpr int WN   = BN / (16 * TN);
    constexpr int NTHR = WM * WN * 32;
    constexpr int KT   = 64;                 // K elements per LDS tile
    constexpr int AI   = (BM * (KT / 8)) / NTHR;  // 16B chunks/thread for A tile
    constexpr int BI   = (BN * (KT / 8)) / NTHR;  // 16B chunks/thread for B tile

    __shared__ __align__(16) __bf16 sA[2][BM * KT];
    __shared__ __align__(16) __bf16 sB[2][BN * KT];

    const int tid  = threadIdx.x;
    const int lane = tid & 31;
    const int wave = tid >> 5;
    const int wm   = (wave % WM) * 32;
    const int wn   = (wave / WM) * (16 * TN);
    const int m0   = blockIdx.x * BM;
    const int n0   = blockIdx.y * BN;
    const int l15  = lane & 15;
    const int hi   = lane >> 4;

    // Per-thread staging assignments (hoisted out of the K loop)
    const __bf16* gA[AI]; unsigned lA[AI];
#pragma unroll
    for (int i = 0; i < AI; ++i) {
        int ch = tid + i * NTHR, row = ch >> 3, seg = ch & 7;
        gA[i] = A + (size_t)(m0 + row) * lda + seg * 8;
        lA[i] = row * KT + seg * 8;
    }
    const __bf16* gB[BI]; unsigned lB[BI];
#pragma unroll
    for (int i = 0; i < BI; ++i) {
        int ch = tid + i * NTHR, col = ch >> 3, seg = ch & 7;
        gB[i] = B + (size_t)(n0 + col) * ldb + seg * 8;
        lB[i] = col * KT + seg * 8;
    }

    auto stage = [&](int buf, int k0) {
#pragma unroll
        for (int i = 0; i < AI; ++i) async_copy16(gA[i] + k0, &sA[buf][lA[i]]);
#pragma unroll
        for (int i = 0; i < BI; ++i) async_copy16(gB[i] + k0, &sB[buf][lB[i]]);
    };

    v8f acc[2][TN] = {};

    const int nTiles = K / KT;
    stage(0, 0);
    wait_async0();
    __syncthreads();

    for (int t = 0; t < nTiles; ++t) {
        const int cur = t & 1;
        if (t + 1 < nTiles) stage(cur ^ 1, (t + 1) * KT);   // async prefetch

        const __bf16* sa = sA[cur];
        const __bf16* sb = sB[cur];
#pragma unroll
        for (int ksub = 0; ksub < 2; ++ksub) {
            // A frag: lanes 0-15 row m, K 0..15; lanes 16-31 K 16..31
            FragB16 a[2];
#pragma unroll
            for (int ti = 0; ti < 2; ++ti) {
                int row = wm + ti * 16 + l15;
                a[ti].q[0] = *(const uint4*)&sa[row * KT + ksub * 32 + hi * 16];
                a[ti].q[1] = *(const uint4*)&sa[row * KT + ksub * 32 + hi * 16 + 8];
            }
            // B frag: lanes 0-15 col n K 0..15; lanes 16-31 K 16..31
            FragB16 b[TN];
#pragma unroll
            for (int tj = 0; tj < TN; ++tj) {
                int col = wn + tj * 16 + l15;
                b[tj].q[0] = *(const uint4*)&sb[col * KT + ksub * 32 + hi * 16];
                b[tj].q[1] = *(const uint4*)&sb[col * KT + ksub * 32 + hi * 16 + 8];
            }
#pragma unroll
            for (int ti = 0; ti < 2; ++ti)
#pragma unroll
                for (int tj = 0; tj < TN; ++tj)
                    acc[ti][tj] = __builtin_amdgcn_wmma_f32_16x16x32_bf16(
                        false, a[ti].v, false, b[tj].v, (short)0, acc[ti][tj], false, false);
        }
        wait_async0();       // prefetched tile landed
        __syncthreads();
    }

    // Epilogue: lane holds col (lane&15); rows r + 8*hi are contiguous -> b128 stores
#pragma unroll
    for (int ti = 0; ti < 2; ++ti)
#pragma unroll
        for (int tj = 0; tj < TN; ++tj) {
            int col = n0 + wn + tj * 16 + l15;
            int row = m0 + wm + ti * 16 + 8 * hi;
            float* cp = C + (size_t)col * ldc + row;
            v8f v = acc[ti][tj];
            *(float4*)cp       = make_float4(v[0], v[1], v[2], v[3]);
            *(float4*)(cp + 4) = make_float4(v[4], v[5], v[6], v[7]);
        }
}

// ---------------------------------------------------------------------------
// Mixing stage 1: n1[nt,c,w] = relu(sum_{k,v} C1[k*2048+c, nt*12+v] *
//                 (distb+distc)[nt,k,v,w]); write bf16 B2 column-major.
// ---------------------------------------------------------------------------
__global__ void k_mix1(const float* __restrict__ C1, const float* __restrict__ dist,
                       __bf16* __restrict__ B2) {
    const int nt = blockIdx.x;
    __shared__ float ds[288];             // (distb + distc)[k][v][w]
    const float* dp = dist + (size_t)nt * 576;
    for (int i = threadIdx.x; i < 288; i += blockDim.x)
        ds[i] = dp[288 + i] + dp[i];      // a=1 (distb) + a=0 (distc)
    __syncthreads();
    for (int c = threadIdx.x; c < 2048; c += blockDim.x) {
        float a[2][12];
#pragma unroll
        for (int k = 0; k < 2; ++k)
#pragma unroll
            for (int v = 0; v < 12; ++v)
                a[k][v] = C1[(size_t)(nt * 12 + v) * 4096 + k * 2048 + c];
#pragma unroll
        for (int w = 0; w < 12; ++w) {
            float s = 0.f;
#pragma unroll
            for (int k = 0; k < 2; ++k)
#pragma unroll
                for (int v = 0; v < 12; ++v)
                    s += a[k][v] * ds[k * 144 + v * 12 + w];
            s = s > 0.f ? s : 0.f;
            B2[(size_t)(nt * 12 + w) * 2048 + c] = (__bf16)s;
        }
    }
}

// ---------------------------------------------------------------------------
// Mixing stage 2 + pooling: produces pooled[nt, 3072] in bf16
// ---------------------------------------------------------------------------
__global__ void k_mix2(const float* __restrict__ C2, const float* __restrict__ dist,
                       __bf16* __restrict__ pooled) {
    const int nt = blockIdx.x;
    __shared__ float db[288], dc[288];
    const float* dp = dist + (size_t)nt * 576;
    for (int i = threadIdx.x; i < 288; i += blockDim.x) {
        db[i] = dp[288 + i];
        dc[i] = dp[i];
    }
    __syncthreads();
    for (int c = threadIdx.x; c < 1024; c += blockDim.x) {
        float a[2][12];
#pragma unroll
        for (int k = 0; k < 2; ++k)
#pragma unroll
            for (int v = 0; v < 12; ++v)
                a[k][v] = C2[(size_t)(nt * 12 + v) * 2048 + k * 1024 + c];
        float left = 0.f, right = 0.f, allf = 0.f;
#pragma unroll
        for (int w = 0; w < 12; ++w) {
            float sb = 0.f, sc = 0.f;
#pragma unroll
            for (int k = 0; k < 2; ++k)
#pragma unroll
                for (int v = 0; v < 12; ++v) {
                    float av = a[k][v];
                    sb += av * db[k * 144 + v * 12 + w];
                    sc += av * dc[k * 144 + v * 12 + w];
                }
            sb = sb > 0.f ? sb : 0.f;
            sc = sc > 0.f ? sc : 0.f;
            if (w < 6) left += sb; else right += sb;
            allf += sc;
        }
        size_t base = (size_t)nt * 3072;
        pooled[base + c]        = (__bf16)(left  * (1.f / 6.f));
        pooled[base + 1024 + c] = (__bf16)(right * (1.f / 6.f));
        pooled[base + 2048 + c] = (__bf16)(allf  * (1.f / 12.f));
    }
}

// ---------------------------------------------------------------------------
// Misc small kernels
// ---------------------------------------------------------------------------
__global__ void k_bias(const float* __restrict__ b_ih, const float* __restrict__ b_hh,
                       float* __restrict__ bsum) {
    int i = blockIdx.x * blockDim.x + threadIdx.x;
    if (i < 8192) bsum[i] = b_ih[i] + b_hh[i];
}

__global__ void k_init_state(__bf16* __restrict__ hbuf, float* __restrict__ cbuf) {
    int i = blockIdx.x * blockDim.x + threadIdx.x;
    if (i < 32 * 2048) hbuf[i] = (__bf16)0.f;
    if (i < 20 * 2048) cbuf[i] = 0.f;
}

// LSTM pointwise update for one sequence step (torch gate order i,f,g,o)
__global__ void k_lstm_point(const float* __restrict__ gx, const float* __restrict__ gh,
                             const float* __restrict__ bsum, float* __restrict__ cbuf,
                             __bf16* __restrict__ hbuf, float* __restrict__ vfeat, int step) {
    int idx = blockIdx.x * blockDim.x + threadIdx.x;   // 20*2048
    int t = idx >> 11, j = idx & 2047;
    const float* gxp = gx + (size_t)(step * 20 + t) * 8192;
    const float* ghp = gh + (size_t)t * 8192;
    float ig = gxp[j]        + ghp[j]        + bsum[j];
    float fg = gxp[2048 + j] + ghp[2048 + j] + bsum[2048 + j];
    float gg = gxp[4096 + j] + ghp[4096 + j] + bsum[4096 + j];
    float og = gxp[6144 + j] + ghp[6144 + j] + bsum[6144 + j];
    float si = 1.f / (1.f + __expf(-ig));
    float sf = 1.f / (1.f + __expf(-fg));
    float so = 1.f / (1.f + __expf(-og));
    float c  = sf * cbuf[idx] + si * tanhf(gg);
    cbuf[idx] = c;
    float h = so * tanhf(c);
    vfeat[(size_t)(step * 20 + t) * 2048 + j] = h;
    hbuf[(size_t)t * 2048 + j] = (__bf16)h;
}

// group_cls[n, cls] = vfeat[n, T-1, :] . Wc[cls, :] + bc[cls]
__global__ void k_cls(const float* __restrict__ vfeat, const float* __restrict__ Wc,
                      const float* __restrict__ bc, float* __restrict__ out) {
    int i = threadIdx.x;          // 256 = 32 n * 8 cls
    int n = i >> 3, cls = i & 7;
    const float* f = vfeat + (size_t)(n * 20 + 19) * 2048;
    const float* w = Wc + (size_t)cls * 2048;
    float s = bc[cls];
    for (int j = 0; j < 2048; ++j) s += f[j] * w[j];
    out[n * 8 + cls] = s;
}

// ---------------------------------------------------------------------------
// Host launch
// ---------------------------------------------------------------------------
extern "C" void kernel_launch(void* const* d_in, const int* in_sizes, int n_in,
                              void* d_out, int out_size, void* d_ws, size_t ws_size,
                              hipStream_t stream) {
    const float* base_out = (const float*)d_in[0];
    const float* dist     = (const float*)d_in[1];
    const float* W1       = (const float*)d_in[2];
    const float* W2       = (const float*)d_in[3];
    const float* w_ih     = (const float*)d_in[4];
    const float* w_hh     = (const float*)d_in[5];
    const float* b_ih     = (const float*)d_in[6];
    const float* b_hh     = (const float*)d_in[7];
    const float* Wc       = (const float*)d_in[8];
    const float* bc       = (const float*)d_in[9];
    float* out = (float*)d_out;

    char* ws = (char*)d_ws;
    size_t off = 0;
    auto alloc = [&](size_t bytes) { void* p = ws + off; off += (bytes + 255) & ~(size_t)255; return p; };
    __bf16* W1b   = (__bf16*)alloc((size_t)4096 * 4096 * 2);
    __bf16* W2b   = (__bf16*)alloc((size_t)2048 * 2048 * 2);
    __bf16* wihb  = (__bf16*)alloc((size_t)8192 * 3072 * 2);
    __bf16* whhb  = (__bf16*)alloc((size_t)8192 * 2048 * 2);
    __bf16* B1    = (__bf16*)alloc((size_t)7680 * 4096 * 2);
    float*  C1    = (float* )alloc((size_t)4096 * 7680 * 4);   // reused as C2
    float*  C2    = C1;
    __bf16* B2    = (__bf16*)alloc((size_t)7680 * 2048 * 2);
    __bf16* pooled= (__bf16*)alloc((size_t)640 * 3072 * 2);
    float*  C3    = (float* )alloc((size_t)8192 * 640 * 4);
    float*  gh    = (float* )alloc((size_t)8192 * 32 * 4);
    __bf16* hbuf  = (__bf16*)alloc((size_t)32 * 2048 * 2);
    float*  cbuf  = (float* )alloc((size_t)20 * 2048 * 4);
    float*  bsum  = (float* )alloc((size_t)8192 * 4);

    auto cvt = [&](const float* in, __bf16* o, int n) {
        k_cvt_bf16<<<(n / 4 + 255) / 256, 256, 0, stream>>>(in, o, n);
    };
    cvt(W1, W1b, 4096 * 4096);
    cvt(base_out, B1, 7680 * 4096);     // base_out rows are already K-contiguous columns
    cvt(W2, W2b, 2048 * 2048);
    cvt(w_ih, wihb, 8192 * 3072);
    cvt(w_hh, whhb, 8192 * 2048);
    k_bias<<<32, 256, 0, stream>>>(b_ih, b_hh, bsum);
    k_init_state<<<(32 * 2048 + 255) / 256, 256, 0, stream>>>(hbuf, cbuf);

    // GEMM1: node1 = W1 @ base_out^T  -> C1[4096, 7680] col-major
    k_gemm_bf16<128, 64, 4><<<dim3(4096 / 128, 7680 / 64), 128, 0, stream>>>(
        W1b, 4096, B1, 4096, C1, 4096, 4096);
    k_mix1<<<640, 256, 0, stream>>>(C1, dist, B2);

    // GEMM2: node2 = W2 @ n1  -> C2[2048, 7680]
    k_gemm_bf16<128, 64, 4><<<dim3(2048 / 128, 7680 / 64), 128, 0, stream>>>(
        W2b, 2048, B2, 2048, C2, 2048, 2048);
    k_mix2<<<640, 256, 0, stream>>>(C2, dist, pooled);

    // GEMM3: gates_x = w_ih @ pooled^T -> C3[8192, 640]
    k_gemm_bf16<128, 64, 4><<<dim3(8192 / 128, 640 / 64), 128, 0, stream>>>(
        wihb, 3072, pooled, 3072, C3, 8192, 3072);

    // LSTM: 32 sequential steps; h padded to 32 columns (cols 20..31 stay zero)
    for (int step = 0; step < 32; ++step) {
        k_gemm_bf16<128, 32, 2><<<dim3(8192 / 128, 1), 128, 0, stream>>>(
            whhb, 2048, hbuf, 2048, gh, 8192, 2048);
        k_lstm_point<<<160, 256, 0, stream>>>(C3, gh, bsum, cbuf, hbuf, out + 256, step);
    }

    k_cls<<<1, 256, 0, stream>>>(out + 256, Wc, bc, out);
}
